// EllEssTeeEmm_70128226009390
// MI455X (gfx1250) — matrix operand order repaired and checked by reading, not compiled, not run on hardware
//
#include <hip/hip_runtime.h>
#include <hip/hip_bf16.h>

typedef __attribute__((ext_vector_type(16))) _Float16 v16h;
typedef __attribute__((ext_vector_type(8)))  _Float16 v8h;
typedef __attribute__((ext_vector_type(8)))  float    v8f;

namespace {
constexpr int LL  = 512;          // sequence length
constexpr int BB  = 64;           // batch
constexpr int IN0 = 256;          // layer-0 input size
constexpr int HH  = 512;          // hidden size
constexpr int G4  = 4 * HH;       // 2048 gate columns (i,f,g,o)
constexpr int LBR = LL * BB;      // 32768 time-major rows
constexpr int SHP = HH + 16;      // padded LDS row stride in f16 (bank-conflict pad)
}

// ---------------------------------------------------------------------------
// f32 -> f16 elementwise convert (weights)
// ---------------------------------------------------------------------------
__global__ void k_f32_to_f16(const float* __restrict__ in, _Float16* __restrict__ out, int n) {
  int i = blockIdx.x * blockDim.x + threadIdx.x;
  if (i < n) out[i] = (_Float16)in[i];
}

// combined bias = bih + bhh (kept f32)
__global__ void k_bias_add(const float* __restrict__ a, const float* __restrict__ b,
                           float* __restrict__ out, int n) {
  int i = blockIdx.x * blockDim.x + threadIdx.x;
  if (i < n) out[i] = a[i] + b[i];
}

// x [B,L,I] f32 -> time-major f16 [(t*B+b), I]
__global__ void k_x_to_f16(const float* __restrict__ x, _Float16* __restrict__ out, int n) {
  int i = blockIdx.x * blockDim.x + threadIdx.x;
  if (i >= n) return;
  int k = i % IN0;
  int r = i / IN0;       // t*BB + b
  int b = r % BB;
  int t = r / BB;
  out[i] = (_Float16)x[(size_t)b * (LL * IN0) + (size_t)t * IN0 + k];
}

// ---------------------------------------------------------------------------
// Input-projection GEMM:  GxT[col][row] = A[row,:] . W[col,:] + bias[col]
//   A   : time-major activations, f16, [LBR x K] row-major
//   W   : f16 [2048 x K] row-major (PyTorch Wih layout -> B matrix = W.T)
//   GxT : f16 [2048][LBR]  (transposed so per-step gate reads are b128s)
// One wave per 16(M)x64(N) tile: A fragment reused across 4 N-tiles.
// ---------------------------------------------------------------------------
__global__ __launch_bounds__(128) void k_gemm_inp(
    const _Float16* __restrict__ A, int K,
    const _Float16* __restrict__ W,
    const float* __restrict__ bias,
    _Float16* __restrict__ GxT) {
  const int wave = threadIdx.x >> 5;
  const int lane = threadIdx.x & 31;
  const int n  = lane & 15;     // D col / A row (lane%16)
  const int h2 = lane >> 4;     // lane half
  const int mt = blockIdx.x * 4 + wave;   // 16-row M tile
  const int ng = blockIdx.y;              // 64-col N group

  const _Float16* Arow = A + (size_t)(mt * 16 + n) * K;
  v8f acc[4] = {};
  for (int k0 = 0; k0 < K; k0 += 32) {
    v8h lo = *(const v8h*)(Arow + k0 + 16 * h2);
    v8h hi = *(const v8h*)(Arow + k0 + 16 * h2 + 8);
    v16h a;
#pragma unroll
    for (int i = 0; i < 8; ++i) { a[i] = lo[i]; a[i + 8] = hi[i]; }
#pragma unroll
    for (int i = 0; i < 4; ++i) {
      const v16h b = *(const v16h*)(W + (size_t)(ng * 64 + i * 16 + n) * K + k0 + 16 * h2);
      acc[i] = __builtin_amdgcn_wmma_f32_16x16x32_f16(false, a, false, b,
                                                      (short)0, acc[i], false, false);
    }
  }
#pragma unroll
  for (int i = 0; i < 4; ++i) {
    const int col = ng * 64 + i * 16 + n;
    const float bs = bias[col];
    v8h hv;
#pragma unroll
    for (int v = 0; v < 8; ++v) hv[v] = (_Float16)(acc[i][v] + bs);
    *(v8h*)(GxT + (size_t)col * LBR + mt * 16 + 8 * h2) = hv;   // 16B contiguous
  }
}

// ---------------------------------------------------------------------------
// Persistent per-layer recurrent kernel. grid.x = 2 (dir 0 fwd, dir 1 rev),
// 1024 threads = 32 waves per direction. Wave w owns hidden cols w*16..+15;
// per M-tile (16 batch rows) it holds one accumulator per gate, so the full
// i/f/g/o tile is register-resident for the fused cell update. h lives in
// padded LDS (f16), c stays in registers in the WMMA C/D layout.
// ---------------------------------------------------------------------------
__global__ __launch_bounds__(1024, 1) void k_lstm_layer(
    const _Float16* __restrict__ GxT0, const _Float16* __restrict__ GxT1,
    const _Float16* __restrict__ Whh0, const _Float16* __restrict__ Whh1,
    _Float16* __restrict__ XS16,   // layer output f16 [(t*B+b) x 2H] (next layer A)
    float* __restrict__ outF,      // final output f32 [B][L][2H]
    int is_last) {
  extern __shared__ _Float16 hsh[];        // [BB][SHP]
  const int dir = blockIdx.x;
  const _Float16* __restrict__ GxT = dir ? GxT1 : GxT0;
  const _Float16* __restrict__ Whh = dir ? Whh1 : Whh0;

  const int tid  = threadIdx.x;
  const int wave = tid >> 5;
  const int lane = tid & 31;
  const int n  = lane & 15;
  const int h2 = lane >> 4;
  const int j0 = wave * 16;                // hidden-column group

  for (int i = tid; i < BB * SHP; i += 1024) hsh[i] = (_Float16)0.f;
  v8f cst[4] = {};                         // cell state, WMMA C/D layout
  __syncthreads();

  for (int t = 0; t < LL; ++t) {
    const int tq = dir ? (LL - 1 - t) : t;
    v8f acc[4][4] = {};                    // [Mtile][gate]
    for (int k0 = 0; k0 < HH; k0 += 32) {
      v16h bfr[4];
#pragma unroll
      for (int g = 0; g < 4; ++g)          // B frag: Whh[col][k] row-major, 32B load
        bfr[g] = *(const v16h*)(Whh + (size_t)(g * HH + j0 + n) * HH + k0 + 16 * h2);
#pragma unroll
      for (int mt = 0; mt < 4; ++mt) {
        const _Float16* ap = hsh + (mt * 16 + n) * SHP + k0;   // h_prev from LDS
        v8h lo = *(const v8h*)(ap + 16 * h2);
        v8h hi = *(const v8h*)(ap + 16 * h2 + 8);
        v16h a;
#pragma unroll
        for (int i = 0; i < 8; ++i) { a[i] = lo[i]; a[i + 8] = hi[i]; }
#pragma unroll
        for (int g = 0; g < 4; ++g)
          acc[mt][g] = __builtin_amdgcn_wmma_f32_16x16x32_f16(false, a, false, bfr[g],
                                                              (short)0, acc[mt][g],
                                                              false, false);
      }
    }
    __syncthreads();   // everyone done READING h_prev before anyone writes h_t

#pragma unroll
    for (int mt = 0; mt < 4; ++mt) {
      const int rbase = tq * BB + mt * 16 + 8 * h2;   // Gx row base (contiguous over v)
      v8h gi = *(const v8h*)(GxT + (size_t)(0 * HH + j0 + n) * LBR + rbase);
      v8h gf = *(const v8h*)(GxT + (size_t)(1 * HH + j0 + n) * LBR + rbase);
      v8h gg = *(const v8h*)(GxT + (size_t)(2 * HH + j0 + n) * LBR + rbase);
      v8h go = *(const v8h*)(GxT + (size_t)(3 * HH + j0 + n) * LBR + rbase);
#pragma unroll
      for (int v = 0; v < 8; ++v) {
        float xi = acc[mt][0][v] + (float)gi[v];
        float xf = acc[mt][1][v] + (float)gf[v];
        float xg = acc[mt][2][v] + (float)gg[v];
        float xo = acc[mt][3][v] + (float)go[v];
        float si = 1.f / (1.f + __expf(-xi));
        float sf = 1.f / (1.f + __expf(-xf));
        float so = 1.f / (1.f + __expf(-xo));
        float e2 = __expf(2.f * xg);
        float tg = (e2 - 1.f) / (e2 + 1.f);
        float cn = sf * cst[mt][v] + si * tg;
        cst[mt][v] = cn;
        float ec = __expf(2.f * cn);
        float th = (ec - 1.f) / (ec + 1.f);
        float hv = so * th;
        const int m = mt * 16 + 8 * h2 + v;           // batch row
        hsh[m * SHP + j0 + n] = (_Float16)hv;
        if (is_last)
          outF[(size_t)m * (LL * 2 * HH) + (size_t)tq * (2 * HH) + dir * HH + j0 + n] = hv;
        else
          XS16[((size_t)tq * BB + m) * (2 * HH) + dir * HH + j0 + n] = (_Float16)hv;
      }
    }
    __syncthreads();   // h_t visible before next step's GEMM reads
  }
}

// ---------------------------------------------------------------------------
extern "C" void kernel_launch(void* const* d_in, const int* in_sizes, int n_in,
                              void* d_out, int out_size, void* d_ws, size_t ws_size,
                              hipStream_t stream) {
  const float* x      = (const float*)d_in[0];
  const float* Wih_f0 = (const float*)d_in[1];
  const float* Whh_f0 = (const float*)d_in[2];
  const float* bih_f0 = (const float*)d_in[3];
  const float* bhh_f0 = (const float*)d_in[4];
  const float* Wih_r0 = (const float*)d_in[5];
  const float* Whh_r0 = (const float*)d_in[6];
  const float* bih_r0 = (const float*)d_in[7];
  const float* bhh_r0 = (const float*)d_in[8];
  const float* Wih_f1 = (const float*)d_in[9];
  const float* Whh_f1 = (const float*)d_in[10];
  const float* bih_f1 = (const float*)d_in[11];
  const float* bhh_f1 = (const float*)d_in[12];
  const float* Wih_r1 = (const float*)d_in[13];
  const float* Whh_r1 = (const float*)d_in[14];
  const float* bih_r1 = (const float*)d_in[15];
  const float* bhh_r1 = (const float*)d_in[16];

  char* ws = (char*)d_ws;
  size_t off = 0;
  auto alloc = [&](size_t bytes) -> char* {
    char* p = ws + off;
    off = (off + bytes + 255) & ~(size_t)255;
    return p;
  };

  _Float16* wihF0h = (_Float16*)alloc((size_t)G4 * IN0 * 2);
  _Float16* whhF0h = (_Float16*)alloc((size_t)G4 * HH * 2);
  _Float16* wihR0h = (_Float16*)alloc((size_t)G4 * IN0 * 2);
  _Float16* whhR0h = (_Float16*)alloc((size_t)G4 * HH * 2);
  _Float16* wihF1h = (_Float16*)alloc((size_t)G4 * 2 * HH * 2);
  _Float16* whhF1h = (_Float16*)alloc((size_t)G4 * HH * 2);
  _Float16* wihR1h = (_Float16*)alloc((size_t)G4 * 2 * HH * 2);
  _Float16* whhR1h = (_Float16*)alloc((size_t)G4 * HH * 2);
  float* bF0 = (float*)alloc((size_t)G4 * 4);
  float* bR0 = (float*)alloc((size_t)G4 * 4);
  float* bF1 = (float*)alloc((size_t)G4 * 4);
  float* bR1 = (float*)alloc((size_t)G4 * 4);
  _Float16* X16  = (_Float16*)alloc((size_t)LBR * IN0 * 2);
  _Float16* XS16 = (_Float16*)alloc((size_t)LBR * 2 * HH * 2);
  _Float16* GxT0 = (_Float16*)alloc((size_t)G4 * LBR * 2);
  _Float16* GxT1 = (_Float16*)alloc((size_t)G4 * LBR * 2);

  auto cvt = [&](const float* src, _Float16* dst, int nE) {
    k_f32_to_f16<<<(nE + 255) / 256, 256, 0, stream>>>(src, dst, nE);
  };
  cvt(Wih_f0, wihF0h, G4 * IN0);
  cvt(Whh_f0, whhF0h, G4 * HH);
  cvt(Wih_r0, wihR0h, G4 * IN0);
  cvt(Whh_r0, whhR0h, G4 * HH);
  cvt(Wih_f1, wihF1h, G4 * 2 * HH);
  cvt(Whh_f1, whhF1h, G4 * HH);
  cvt(Wih_r1, wihR1h, G4 * 2 * HH);
  cvt(Whh_r1, whhR1h, G4 * HH);
  k_bias_add<<<(G4 + 255) / 256, 256, 0, stream>>>(bih_f0, bhh_f0, bF0, G4);
  k_bias_add<<<(G4 + 255) / 256, 256, 0, stream>>>(bih_r0, bhh_r0, bR0, G4);
  k_bias_add<<<(G4 + 255) / 256, 256, 0, stream>>>(bih_f1, bhh_f1, bF1, G4);
  k_bias_add<<<(G4 + 255) / 256, 256, 0, stream>>>(bih_r1, bhh_r1, bR1, G4);

  {
    int nE = LL * BB * IN0;
    k_x_to_f16<<<(nE + 255) / 256, 256, 0, stream>>>(x, X16, nE);
  }

  const dim3 gGemm(LBR / 64, G4 / 64);     // (512, 32), 128-thread blocks
  const size_t ldsBytes = (size_t)BB * SHP * 2;

  // ---- layer 0 ----
  k_gemm_inp<<<gGemm, 128, 0, stream>>>(X16, IN0, wihF0h, bF0, GxT0);
  k_gemm_inp<<<gGemm, 128, 0, stream>>>(X16, IN0, wihR0h, bR0, GxT1);
  k_lstm_layer<<<2, 1024, ldsBytes, stream>>>(GxT0, GxT1, whhF0h, whhR0h,
                                              XS16, (float*)d_out, 0);
  // ---- layer 1 ----
  k_gemm_inp<<<gGemm, 128, 0, stream>>>(XS16, 2 * HH, wihF1h, bF1, GxT0);
  k_gemm_inp<<<gGemm, 128, 0, stream>>>(XS16, 2 * HH, wihR1h, bR1, GxT1);
  k_lstm_layer<<<2, 1024, ldsBytes, stream>>>(GxT0, GxT1, whhF1h, whhR1h,
                                              XS16, (float*)d_out, 1);
}
